// MultiHeadAttention_87557203297096
// MI455X (gfx1250) — matrix operand order repaired and checked by reading, not compiled, not run on hardware
//
#include <hip/hip_runtime.h>

// ---- problem constants (match reference) ----
#define Bb   4
#define Ss   2048
#define Dd   1024
#define Hh   16
#define DK   64
#define NEGV (-10000.0f)

typedef __attribute__((ext_vector_type(16))) __bf16 v16bf;
typedef __attribute__((ext_vector_type(8)))  __bf16 v8bf;
typedef __attribute__((ext_vector_type(8)))  float  v8f;
typedef __attribute__((ext_vector_type(4)))  float  f4;
typedef __attribute__((ext_vector_type(4)))  int    i4;

#define STRIDE 2052  // 16B-aligned, bank-conflict-free LDS row stride (floats)

// ---- operand builders ----
// A-operand from fp32: 16 contiguous (chunks [0..7] and [8..15])
__device__ __forceinline__ v16bf cvt16_split_f32(const float* p) {
  v16bf r;
#pragma unroll
  for (int i = 0; i < 8; ++i) { r[i] = (__bf16)p[i]; r[i + 8] = (__bf16)p[i + 8]; }
  return r;
}
// B-operand from fp32: 16 contiguous
__device__ __forceinline__ v16bf cvt16_contig_f32(const float* p) {
  v16bf r;
#pragma unroll
  for (int i = 0; i < 16; ++i) r[i] = (__bf16)p[i];
  return r;
}
// A-operand from bf16 memory: two 16B loads, register concat (no VALU)
__device__ __forceinline__ v16bf ld16_split_bf16(const __bf16* p) {
  v8bf lo = *(const v8bf*)p;
  v8bf hi = *(const v8bf*)(p + 8);
  v16bf r;
#pragma unroll
  for (int i = 0; i < 8; ++i) { r[i] = lo[i]; r[i + 8] = hi[i]; }
  return r;
}

// ---------------------------------------------------------------------------
// Kernel 1: y = x @ W^T + b  (bf16 WMMA, fp32 accumulate, bf16 output)
//   mode 0 (Q,K): transposed tile (A=W rows, B=X^T) -> 8 D-elems contiguous
//                 along head dim -> one 16B bf16 store. Layout [B,H,S,64].
//   mode 1 (V):   normal orientation -> 8 D-elems contiguous along S in the
//                 transposed [B,H,64,S] layout -> one 16B bf16 store.
// grid: (B*S/16, D/128), block 256 (8 waves, one 16x16 tile each)
// ---------------------------------------------------------------------------
__global__ __launch_bounds__(256) void proj_gemm(
    const float* __restrict__ X,    // [B*S, D]
    const float* __restrict__ W,    // [1024, D]  (y = x W^T)
    const float* __restrict__ bias, // [1024]
    __bf16* __restrict__ out,
    int vTranspose)
{
  const int lane = threadIdx.x & 31;
  const int wave = threadIdx.x >> 5;
  const int m0 = blockIdx.x * 16;                 // sample-row tile (b*S+s)
  const int n0 = blockIdx.y * 128 + wave * 16;    // feature tile
  const int r  = lane & 15;
  const int hi = lane >> 4;

  const int bidx   = m0 >> 11;                    // S == 2048
  const int s_base = m0 & 2047;
  const int h      = n0 >> 6;
  const size_t bh  = (size_t)bidx * Hh + h;

  v8f c;
  if (!vTranspose) {
    const float* bp = bias + n0 + hi * 8;
#pragma unroll
    for (int j = 0; j < 8; ++j) c[j] = bp[j];
    const float* arow = W + (size_t)(n0 + r) * Dd;     // A: feature rows of W
    const float* brow = X + (size_t)(m0 + r) * Dd;     // B: col = sample, K contig
#pragma unroll 2
    for (int k0 = 0; k0 < Dd; k0 += 32) {
      v16bf a = cvt16_split_f32(arow + k0 + hi * 16);
      v16bf b = cvt16_contig_f32(brow + k0 + hi * 16);
      c = __builtin_amdgcn_wmma_f32_16x16x32_bf16(false, a, false, b,
                                                  (short)0, c, false, false);
    }
    const int s  = s_base + r;
    const int dv = (n0 & 63) + hi * 8;
    __bf16* dst = out + ((bh * Ss + s) * DK + dv);
    v8bf o;
#pragma unroll
    for (int j = 0; j < 8; ++j) o[j] = (__bf16)c[j];
    *(v8bf*)dst = o;
  } else {
    const float bv = bias[n0 + r];
#pragma unroll
    for (int j = 0; j < 8; ++j) c[j] = bv;
    const float* arow = X + (size_t)(m0 + r) * Dd;     // A: sample rows of X
    const float* brow = W + (size_t)(n0 + r) * Dd;     // B: col = feature, K contig
#pragma unroll 2
    for (int k0 = 0; k0 < Dd; k0 += 32) {
      v16bf a = cvt16_split_f32(arow + k0 + hi * 16);
      v16bf b = cvt16_contig_f32(brow + k0 + hi * 16);
      c = __builtin_amdgcn_wmma_f32_16x16x32_bf16(false, a, false, b,
                                                  (short)0, c, false, false);
    }
    const int dv = (n0 + r) & 63;
    const int s  = s_base + hi * 8;
    __bf16* dst = out + ((bh * DK + dv) * Ss + s);
    v8bf o;
#pragma unroll
    for (int j = 0; j < 8; ++j) o[j] = (__bf16)c[j];
    *(v8bf*)dst = o;
  }
}

// ---------------------------------------------------------------------------
// Kernel 2: fused  scores -> mask+softmax -> attn writeback (NT) -> P@V
// One block per (b, h, 16-query strip). 256 threads / 8 waves.
// LDS: 16 x 2052 f32 score strip + 16x64 f32 output tile (~135 KB -> 2 WG/WGP)
// ---------------------------------------------------------------------------
__global__ __launch_bounds__(256) void attn_kernel(
    const __bf16* __restrict__ qh,  // [B,H,S,64] bf16
    const __bf16* __restrict__ kh,  // [B,H,S,64] bf16
    const __bf16* __restrict__ vt,  // [B,H,64,S] bf16
    const int*    __restrict__ mask,// [B,S,S]
    float* __restrict__ attn,       // [B,H,S,S]
    float* __restrict__ outp)       // [B,S,1024] pre-LN
{
  extern __shared__ float smem[];
  float* strip  = smem;                      // 16 * STRIDE floats
  float* outacc = smem + 16 * STRIDE;        // 16 * 64 floats

  const int tid  = threadIdx.x;
  const int lane = tid & 31;
  const int wave = tid >> 5;
  const int qt = blockIdx.x & 127;
  const int h  = (blockIdx.x >> 7) & 15;
  const int b  = blockIdx.x >> 11;
  const int q0 = qt * 16;
  const int r  = lane & 15;
  const int hi = lane >> 4;

  // zero the cross-wave accumulation tile (used after phase-2 barrier)
  for (int i = tid; i < 16 * 64; i += 256) outacc[i] = 0.0f;

  const size_t bh = (size_t)b * Hh + h;
  const __bf16* __restrict__ qbase = qh + bh * Ss * DK;
  const __bf16* __restrict__ kbase = kh + bh * Ss * DK;
  const __bf16* __restrict__ vbase = vt + bh * DK * Ss;

  // ---- Q strip (16x64) into WMMA A registers (two K=32 slabs), bf16 direct ----
  const __bf16* qrow = qbase + (size_t)(q0 + r) * DK;
  const v16bf a0 = ld16_split_bf16(qrow + hi * 16);
  const v16bf a1 = ld16_split_bf16(qrow + 32 + hi * 16);

  // ---- Phase 1: raw scaled scores for keys [wave*256, wave*256+256) ----
  for (int t = 0; t < 16; ++t) {
    const int n0  = wave * 256 + t * 16;
    const int key = n0 + r;
    const __bf16* krow = kbase + (size_t)key * DK;   // B col = key, K contiguous
    v16bf b0 = *(const v16bf*)(krow + hi * 16);      // one 32-byte load (16 bf16)
    v16bf b1 = *(const v16bf*)(krow + 32 + hi * 16);
    if (t < 15) __builtin_prefetch(krow + 16 * DK, 0, 3);  // global_prefetch_b8
    v8f c = {};
    c = __builtin_amdgcn_wmma_f32_16x16x32_bf16(false, a0, false, b0, (short)0, c, false, false);
    c = __builtin_amdgcn_wmma_f32_16x16x32_bf16(false, a1, false, b1, (short)0, c, false, false);
#pragma unroll
    for (int j = 0; j < 8; ++j)
      strip[(hi * 8 + j) * STRIDE + key] = c[j] * 0.125f;   // dk^-0.5
  }
  __syncthreads();

  // ---- Phase 2: mask adders + softmax (16 threads/row, 128 cols each) ----
  {
    const int row = tid >> 4;
    const int seg = tid & 15;
    const int q   = q0 + row;
    float* rp = strip + row * STRIDE + seg * 128;
    const int* mp = mask + ((size_t)b * Ss + q) * Ss + seg * 128;  // coalesced
    const int key0 = seg * 128;
    float mx = -3.0e38f;
    for (int i = 0; i < 128; i += 4) {
      f4 p  = *(const f4*)(rp + i);
      i4 mk = *(const i4*)(mp + i);
#pragma unroll
      for (int u = 0; u < 4; ++u) {
        float s = p[u] + (1.0f - (float)mk[u]) * NEGV;       // padding mask
        if (key0 + i + u > q) s += NEGV;                     // causal mask
        p[u] = s;
        mx = fmaxf(mx, s);
      }
      *(f4*)(rp + i) = p;
    }
#pragma unroll
    for (int m = 8; m >= 1; m >>= 1) mx = fmaxf(mx, __shfl_xor(mx, m, 16));
    float sum = 0.0f;
    for (int i = 0; i < 128; ++i) { float e = __expf(rp[i] - mx); rp[i] = e; sum += e; }
#pragma unroll
    for (int m = 8; m >= 1; m >>= 1) sum += __shfl_xor(sum, m, 16);
    const float inv = 1.0f / sum;
    float* gattn = attn + (bh * Ss + q) * Ss + seg * 128;
    for (int i = 0; i < 128; i += 4) {      // normalize + stream 1 GiB attn (NT)
      f4 p = *(const f4*)(rp + i);
      p *= inv;
      *(f4*)(rp + i) = p;
      __builtin_nontemporal_store(p, (f4*)(gattn + i));
    }
  }
  __syncthreads();

  // ---- Phase 3: O = P @ V ; wave owns a 256-key range, all 4 dv tiles ----
  // A (P) converted from LDS once per k-step, reused for 4 B-tiles; four
  // independent accumulators break WMMA->WMMA RAW chains.
  {
    const int kbase = wave * 256;
    const __bf16* vrow = vbase + (size_t)r * Ss;     // dv = nt*16 + r
    v8f acc0 = {}, acc1 = {}, acc2 = {}, acc3 = {};
    for (int ks = 0; ks < 8; ++ks) {
      const int k0 = kbase + ks * 32;
      v16bf A = cvt16_split_f32(strip + r * STRIDE + k0 + hi * 16);
      const __bf16* vp = vrow + k0 + hi * 16;
      v16bf B0 = *(const v16bf*)(vp);
      v16bf B1 = *(const v16bf*)(vp + (size_t)16 * Ss);
      v16bf B2 = *(const v16bf*)(vp + (size_t)32 * Ss);
      v16bf B3 = *(const v16bf*)(vp + (size_t)48 * Ss);
      acc0 = __builtin_amdgcn_wmma_f32_16x16x32_bf16(false, A, false, B0, (short)0, acc0, false, false);
      acc1 = __builtin_amdgcn_wmma_f32_16x16x32_bf16(false, A, false, B1, (short)0, acc1, false, false);
      acc2 = __builtin_amdgcn_wmma_f32_16x16x32_bf16(false, A, false, B2, (short)0, acc2, false, false);
      acc3 = __builtin_amdgcn_wmma_f32_16x16x32_bf16(false, A, false, B3, (short)0, acc3, false, false);
    }
    // cross-wave reduce via native LDS float atomics (ds_add_f32)
#pragma unroll
    for (int j = 0; j < 8; ++j) {
      const int m = hi * 8 + j;
      atomicAdd(&outacc[m * 64 +      r], acc0[j]);
      atomicAdd(&outacc[m * 64 + 16 + r], acc1[j]);
      atomicAdd(&outacc[m * 64 + 32 + r], acc2[j]);
      atomicAdd(&outacc[m * 64 + 48 + r], acc3[j]);
    }
  }
  __syncthreads();

  // ---- Phase 4: store pre-LN output tile ----
#pragma unroll
  for (int i = 0; i < 4; ++i) {
    const int idx = tid + i * 256;                   // 16x64 tile
    const int m = idx >> 6, n = idx & 63;
    outp[((size_t)b * Ss + q0 + m) * Dd + h * DK + n] = outacc[idx];
  }
}

// ---------------------------------------------------------------------------
// Kernel 3: out = LayerNorm(out_pre + residual) * g + b   (in-place, 1 row/block)
// ---------------------------------------------------------------------------
__global__ __launch_bounds__(256) void ln_kernel(
    float* __restrict__ out,
    const float* __restrict__ resid,
    const float* __restrict__ g,
    const float* __restrict__ be)
{
  __shared__ float ssum[8], ssum2[8];
  const size_t row = blockIdx.x;
  float* rp = out + row * Dd;
  const float* qp = resid + row * Dd;
  const int tid = threadIdx.x;

  float vals[4], s = 0.0f, s2 = 0.0f;
#pragma unroll
  for (int i = 0; i < 4; ++i) {
    const float v = rp[tid + i * 256] + qp[tid + i * 256];
    vals[i] = v; s += v; s2 += v * v;
  }
#pragma unroll
  for (int m = 16; m >= 1; m >>= 1) { s += __shfl_xor(s, m, 32); s2 += __shfl_xor(s2, m, 32); }
  if ((tid & 31) == 0) { ssum[tid >> 5] = s; ssum2[tid >> 5] = s2; }
  __syncthreads();
  s = 0.0f; s2 = 0.0f;
#pragma unroll
  for (int i = 0; i < 8; ++i) { s += ssum[i]; s2 += ssum2[i]; }
  const float mu   = s * (1.0f / (float)Dd);
  const float var  = s2 * (1.0f / (float)Dd) - mu * mu;
  const float rinv = rsqrtf(var + 1e-5f);
#pragma unroll
  for (int i = 0; i < 4; ++i) {
    const int c = tid + i * 256;
    rp[c] = (vals[i] - mu) * rinv * g[c] + be[c];
  }
}

// ---------------------------------------------------------------------------
extern "C" void kernel_launch(void* const* d_in, const int* in_sizes, int n_in,
                              void* d_out, int out_size, void* d_ws, size_t ws_size,
                              hipStream_t stream) {
  (void)in_sizes; (void)n_in; (void)out_size; (void)ws_size;
  const float* q    = (const float*)d_in[0];
  const float* k    = (const float*)d_in[1];
  const float* v    = (const float*)d_in[2];
  const int*   mask = (const int*)  d_in[3];
  const float* Wq   = (const float*)d_in[4];
  const float* bq   = (const float*)d_in[5];
  const float* Wk   = (const float*)d_in[6];
  const float* bk   = (const float*)d_in[7];
  const float* Wv   = (const float*)d_in[8];
  const float* bv   = (const float*)d_in[9];
  const float* lng  = (const float*)d_in[10];
  const float* lnb  = (const float*)d_in[11];

  float* out  = (float*)d_out;
  float* attn = out + (size_t)Bb * Ss * Dd;          // tuple: (out, attn)

  __bf16* qh = (__bf16*)d_ws;                        // [B,H,S,64]  16 MB
  __bf16* kh = qh + (size_t)Bb * Hh * Ss * DK;       // [B,H,S,64]  16 MB
  __bf16* vt = kh + (size_t)Bb * Hh * Ss * DK;       // [B,H,64,S]  16 MB

  const dim3 blk(256);
  const dim3 g1(Bb * Ss / 16, Dd / 128);             // 512 x 8
  proj_gemm<<<g1, blk, 0, stream>>>(q, Wq, bq, qh, 0);
  proj_gemm<<<g1, blk, 0, stream>>>(k, Wk, bk, kh, 0);
  proj_gemm<<<g1, blk, 0, stream>>>(v, Wv, bv, vt, 1);

  const size_t lds2 = (size_t)(16 * STRIDE + 16 * 64) * sizeof(float); // ~135 KB (2 WG / 320 KB WGP)
  attn_kernel<<<dim3(Bb * Hh * (Ss / 16)), blk, lds2, stream>>>(qh, kh, vt, mask, attn, out);

  ln_kernel<<<dim3(Bb * Ss), blk, 0, stream>>>(out, q, lng, lnb);
}
